// PerformerAttention_86414741996184
// MI455X (gfx1250) — matrix operand order repaired and checked by reading, not compiled, not run on hardware
//
#include <hip/hip_runtime.h>

// ---------------------------------------------------------------------------
// Performer causal linear attention, CDNA5 (gfx1250) WMMA implementation.
// B=4 S=2048 D=768 H=12 HD=64 F=266 (padded to 320), chunk size C=128.
// - All matmuls: v_wmma_f32_16x16x32_bf16, f32 accumulate.
// - Every fragment load is a contiguous 128-bit load (B operands pre-
//   transposed to K-major storage; A operands row-major).
// - V chunks are DMA'd global->LDS with the Tensor Data Mover (TDM),
//   overlapped with WMMA work, then transposed in place in LDS.
// ---------------------------------------------------------------------------

typedef __bf16 bf16;
typedef __bf16 v16bf  __attribute__((ext_vector_type(16)));
typedef __bf16 bf16x8 __attribute__((ext_vector_type(8)));
typedef float  v8f    __attribute__((ext_vector_type(8)));
typedef unsigned int u32x4 __attribute__((ext_vector_type(4)));
typedef int          i32x4 __attribute__((ext_vector_type(4)));
typedef int          i32x8 __attribute__((ext_vector_type(8)));

constexpr int   kB = 4, kS = 2048, kD = 768, kH = 12, kHD = 64;
constexpr int   kFreal = 266, kFP = 320;      // feature dim padded with zeros
constexpr int   kC = 128, kNC = kS / kC;      // causal chunks
constexpr int   kM = kB * kS;                 // 8192 token rows
constexpr float kScale = 0.125f;              // HD^-0.5
constexpr float kEps   = 1e-6f;

#if defined(__has_builtin)
#if __has_builtin(__builtin_amdgcn_tensor_load_to_lds)
#define PERF_HAVE_TDM 1
#endif
#endif
#ifndef PERF_HAVE_TDM
#define PERF_HAVE_TDM 0
#endif

__device__ __forceinline__ v8f wmma_bf(v16bf a, v16bf b, v8f c) {
  return __builtin_amdgcn_wmma_f32_16x16x32_bf16(false, a, false, b, (short)0, c,
                                                 false, false);
}

// A fragment: A row-major [*, lda]; caller passes row = m0 + (lane&15).
__device__ __forceinline__ v16bf frag_a(const bf16* A, int lda, int row, int k0,
                                        int lane) {
  const int off = (lane & 16) ? 8 : 0;
  const bf16* p = A + (size_t)row * lda + (k0 + off);
  bf16x8 lo = *(const bf16x8*)p;
  bf16x8 hi = *(const bf16x8*)(p + 16);
  v16bf a;
#pragma unroll
  for (int i = 0; i < 8; ++i) { a[i] = lo[i]; a[i + 8] = hi[i]; }
  return a;
}

// B fragment from K-major (transposed) storage Bt[col][k]; caller passes
// col = n0 + (lane&15).
__device__ __forceinline__ v16bf frag_b(const bf16* Bt, int ldbt, int col, int k0,
                                        int lane) {
  const int off = (lane & 16) ? 8 : 0;
  const bf16* p = Bt + (size_t)col * ldbt + (k0 + off);
  bf16x8 lo = *(const bf16x8*)p;
  bf16x8 hi = *(const bf16x8*)(p + 16);
  v16bf b;
#pragma unroll
  for (int i = 0; i < 8; ++i) { b[i] = lo[i]; b[i + 8] = hi[i]; }
  return b;
}

// ---------------- Tensor Data Mover: 2D bf16 tile -> LDS -------------------
// Builds D# per CDNA5 ISA 8.3/8.4: group0 {count=1, lds_addr, global_addr,
// type=2}; group1 {data_size=2B, tensor dims, tile dims, dim0 stride}.
__device__ __forceinline__ unsigned int lds_offset_of(const void* p) {
  return (unsigned int)(unsigned long long)p;  // addr[31:0] == LDS offset
}

#if PERF_HAVE_TDM
__device__ __forceinline__ void tdm_load_2d_bf16(const void* gptr,
                                                 unsigned int lds_addr,
                                                 unsigned int row_elems,
                                                 unsigned int rows,
                                                 unsigned int stride_elems) {
  unsigned long long ga = (unsigned long long)gptr;
  u32x4 g0;
  g0[0] = 1u;                                        // count=1 (valid user D#)
  g0[1] = lds_addr;                                  // lds_addr (bytes)
  g0[2] = (unsigned int)ga;                          // global_addr[31:0]
  g0[3] = (unsigned int)(ga >> 32) | (2u << 30);     // addr[56:32] | type=2
  i32x8 g1;
  g1[0] = (int)(1u << 16);                           // data_size=1 (2 bytes)
  g1[1] = (int)((row_elems & 0xFFFFu) << 16);        // tensor_dim0[15:0]
  g1[2] = (int)(((row_elems >> 16) & 0xFFFFu) |      // tensor_dim0[31:16]
                ((rows & 0xFFFFu) << 16));           // tensor_dim1[15:0]
  g1[3] = (int)(((rows >> 16) & 0xFFFFu) |           // tensor_dim1[31:16]
                ((row_elems & 0xFFFFu) << 16));      // tile_dim0
  g1[4] = (int)(rows & 0xFFFFu);                     // tile_dim1 (tile_dim2=0)
  g1[5] = (int)stride_elems;                         // tensor_dim0_stride lo
  g1[6] = 0;                                         // stride hi / dim1 stride
  g1[7] = 0;
  i32x4 z4 = {0, 0, 0, 0};
#if __clang_major__ >= 23
  i32x8 z8 = {0, 0, 0, 0, 0, 0, 0, 0};
  __builtin_amdgcn_tensor_load_to_lds(g0, g1, z4, z4, z8, 0);
#else
  __builtin_amdgcn_tensor_load_to_lds(g0, g1, z4, z4, 0);
#endif
}
__device__ __forceinline__ void tdm_wait0() {
#if __has_builtin(__builtin_amdgcn_s_wait_tensorcnt)
  __builtin_amdgcn_s_wait_tensorcnt(0);
#else
  asm volatile("s_wait_tensorcnt 0x0" ::: "memory");
#endif
}
#endif

// In-place LDS transpose of a 128x64 bf16 tile (row layout [s][64]) into
// column layout [j][128], via 32 registers/thread. Caller barriers before.
__device__ __forceinline__ void transpose_v_inplace(unsigned short* vbuf,
                                                    int tid) {
  unsigned short r[32];
#pragma unroll
  for (int i = 0; i < 32; ++i) r[i] = vbuf[tid + 256 * i];
  __syncthreads();
#pragma unroll
  for (int i = 0; i < 32; ++i) {
    int idx = tid + 256 * i;
    vbuf[(idx & 63) * kC + (idx >> 6)] = r[i];
  }
  __syncthreads();
}

// --------------------------- prep kernels ----------------------------------

__global__ void perf_cvt_bf16(const float* __restrict__ s, bf16* __restrict__ d,
                              int n) {
  int i = blockIdx.x * 256 + threadIdx.x;
  if (i < n) d[i] = (bf16)s[i];
}

__global__ void perf_transpose_cvt(const float* __restrict__ W,
                                   bf16* __restrict__ Wt) {
  int i = blockIdx.x * 256 + threadIdx.x;
  if (i >= kD * kD) return;
  int n = i / kD, k = i % kD;
  Wt[i] = (bf16)W[(size_t)k * kD + n];
}

// Fused projection: Wt[h][f][k] = scale * sum_j W[k][h*64+j] * P[h][f][j]
__global__ void perf_fuse_proj(const float* __restrict__ W,
                               const float* __restrict__ P,
                               bf16* __restrict__ Wt, float scale) {
  int i = blockIdx.x * 256 + threadIdx.x;
  if (i >= kH * kFP * kD) return;
  int k = i % kD;
  int f = (i / kD) % kFP;
  int h = i / (kD * kFP);
  float acc = 0.f;
  if (f < kFreal) {
    const float* wc = W + (size_t)k * kD + h * kHD;
    const float* pr = P + ((size_t)h * kFreal + f) * kHD;
#pragma unroll 8
    for (int j = 0; j < kHD; ++j) acc += wc[j] * pr[j];
    acc *= scale;
  }
  Wt[i] = (bf16)acc;
}

// --------------------- projection GEMM (Q'/K'/V) ---------------------------
__global__ __launch_bounds__(256) void perf_proj_gemm(
    const bf16* __restrict__ A, const bf16* __restrict__ Btbase, int ldbt,
    long headBtStride, int col0mult, bf16* __restrict__ outp, int ldo,
    int Nvalid, int Kdim, int do_relu) {
  const int h = blockIdx.z;
  const int tid = threadIdx.x;
  const int lane = tid & 31, l15 = lane & 15, hi = (lane >> 4) & 1;
  const int w = __builtin_amdgcn_readfirstlane(tid >> 5);
  const int m0 = blockIdx.x * 64 + (w & 3) * 16;
  const int nbase = blockIdx.y * 128 + (w >> 2) * 64;
  if (nbase >= Nvalid) return;
  const bf16* Bt = Btbase + (size_t)h * headBtStride;
  const int col0 = h * col0mult;
  const int arow = m0 + l15;

  v8f acc[4] = {};
  for (int k0 = 0; k0 < Kdim; k0 += 32) {
    __builtin_prefetch(A + (size_t)arow * Kdim + k0 + 128, 0, 1);
    v16bf a = frag_a(A, Kdim, arow, k0, lane);
    v16bf bb[4];
#pragma unroll
    for (int nt = 0; nt < 4; ++nt)
      bb[nt] = frag_b(Bt, ldbt, col0 + nbase + nt * 16 + l15, k0, lane);
#pragma unroll
    for (int nt = 0; nt < 4; ++nt) acc[nt] = wmma_bf(a, bb[nt], acc[nt]);
  }
#pragma unroll
  for (int nt = 0; nt < 4; ++nt) {
    int coln = nbase + nt * 16 + l15;
    if (coln >= Nvalid) continue;
#pragma unroll
    for (int r = 0; r < 8; ++r) {
      int row = m0 + r + 8 * hi;
      int bb2 = row >> 11, ss = row & (kS - 1);
      float v = acc[nt][r];
      if (do_relu) v = v > 0.f ? v : 0.f;
      outp[(((size_t)(bb2 * kH + h) * kS + ss) * ldo) + coln] = (bf16)v;
    }
  }
}

// ----------------- per-chunk state: KV = K'^T V, ksum = colsum(K') ---------
__global__ __launch_bounds__(256) void perf_chunk_kv(
    const bf16* __restrict__ kp, const bf16* __restrict__ v,
    bf16* __restrict__ KVloc, float* __restrict__ ksumloc) {
  const int c = blockIdx.x, h = blockIdx.y, b = blockIdx.z;
  const int bh = b * kH + h, c0 = c * kC;
  const bf16* kp_bh = kp + (size_t)bh * kS * kFP;
  const bf16* v_bh = v + (size_t)bh * kS * kHD;
  __shared__ unsigned short vbuf[kHD * kC];  // V tile: rows then transposed
  __shared__ bf16 kpT[kFP][40];              // K'^T, 32 s-columns at a time
  const int tid = threadIdx.x, lane = tid & 31, l15 = lane & 15;
  const int hi = (lane >> 4) & 1;
  const int wu = __builtin_amdgcn_readfirstlane(tid >> 5);

#if PERF_HAVE_TDM
  if (wu == 0)  // async DMA of V chunk, overlapped with K' staging below
    tdm_load_2d_bf16(v_bh + (size_t)c0 * kHD, lds_offset_of(vbuf), kHD, kC, kHD);
#else
  for (int i = tid; i < kC * kHD; i += 256)
    vbuf[i] = ((const unsigned short*)(v_bh + (size_t)c0 * kHD))[i];
#endif
  // stage K'^T slab 0 while the DMA runs
  for (int i = tid; i < 32 * kFP; i += 256) {
    int s = i / kFP, f = i % kFP;
    kpT[f][s] = kp_bh[(size_t)(c0 + s) * kFP + f];
  }
#if PERF_HAVE_TDM
  if (wu == 0) tdm_wait0();
#endif
  __syncthreads();
  transpose_v_inplace(vbuf, tid);            // vbuf now V^T [64][128]
  const bf16* vT = (const bf16*)vbuf;

  v8f acc[3][4] = {};
  for (int ks = 0; ks < 4; ++ks) {
    if (ks) {
      __syncthreads();
      for (int i = tid; i < 32 * kFP; i += 256) {
        int s = i / kFP, f = i % kFP;
        kpT[f][s] = kp_bh[(size_t)(c0 + ks * 32 + s) * kFP + f];
      }
      __syncthreads();
    }
    v16bf bfr[4];
#pragma unroll
    for (int nt = 0; nt < 4; ++nt)
      bfr[nt] = frag_b(vT, kC, nt * 16 + l15, ks * 32, lane);
#pragma unroll
    for (int t = 0; t < 3; ++t) {
      const int mt = wu + 8 * t;
      if (mt >= kFP / 16) continue;
      v16bf a = frag_a(&kpT[0][0], 40, mt * 16 + l15, 0, lane);
#pragma unroll
      for (int nt = 0; nt < 4; ++nt) acc[t][nt] = wmma_bf(a, bfr[nt], acc[t][nt]);
    }
  }
  bf16* KVc = KVloc + ((size_t)bh * kNC + c) * kFP * kHD;  // [f][j]
#pragma unroll
  for (int t = 0; t < 3; ++t) {
    const int mt = wu + 8 * t;
    if (mt >= kFP / 16) continue;
#pragma unroll
    for (int nt = 0; nt < 4; ++nt)
#pragma unroll
      for (int r = 0; r < 8; ++r)
        KVc[(size_t)(mt * 16 + r + 8 * hi) * kHD + nt * 16 + l15] =
            (bf16)acc[t][nt][r];
  }
  float* ksc = ksumloc + ((size_t)bh * kNC + c) * kFP;
  for (int f = tid; f < kFP; f += 256) {
    float s = 0.f;
    for (int s2 = 0; s2 < kC; ++s2) s += (float)kp_bh[(size_t)(c0 + s2) * kFP + f];
    ksc[f] = s;
  }
}

// ------------- exclusive prefix over chunks (writes K-major KVpre^T) -------
__global__ void perf_prefix_kv(const bf16* __restrict__ KVloc,
                               bf16* __restrict__ KVpreT) {
  size_t idx = (size_t)blockIdx.x * 256 + threadIdx.x;
  if (idx >= (size_t)kB * kH * kFP * kHD) return;
  int j = idx & 63;
  int f = (int)((idx >> 6) % kFP);
  int bh = (int)(idx / ((size_t)kFP * kHD));
  float run = 0.f;
  for (int c = 0; c < kNC; ++c) {
    KVpreT[(((size_t)bh * kNC + c) * kHD + j) * kFP + f] = (bf16)run;
    run += (float)KVloc[(((size_t)bh * kNC + c) * kFP + f) * kHD + j];
  }
}

__global__ void perf_prefix_ksum(const float* __restrict__ ksl,
                                 float* __restrict__ ksp) {
  int idx = blockIdx.x * 256 + threadIdx.x;
  if (idx >= kB * kH * kFP) return;
  int f = idx % kFP, bh = idx / kFP;
  float run = 0.f;
  for (int c = 0; c < kNC; ++c) {
    ksp[((size_t)bh * kNC + c) * kFP + f] = run;
    run += ksl[((size_t)bh * kNC + c) * kFP + f];
  }
}

// --------------- intra-chunk causal attention + combine --------------------
__global__ __launch_bounds__(256) void perf_intra(
    const bf16* __restrict__ qp, const bf16* __restrict__ kp,
    const bf16* __restrict__ v, const bf16* __restrict__ KVpreT,
    const float* __restrict__ ksumpre, bf16* __restrict__ attn) {
  const int c = blockIdx.x, h = blockIdx.y, b = blockIdx.z;
  const int bh = b * kH + h, c0 = c * kC;
  const bf16* qp_bh = qp + (size_t)bh * kS * kFP;
  const bf16* kp_bh = kp + (size_t)bh * kS * kFP;
  const bf16* v_bh = v + (size_t)bh * kS * kHD;
  const bf16* KVt = KVpreT + ((size_t)bh * kNC + c) * kHD * kFP;  // [j][f]
  const float* ksc = ksumpre + ((size_t)bh * kNC + c) * kFP;
  __shared__ bf16 sS[kC][kC + 8];
  __shared__ unsigned short vbuf[kHD * kC];
  __shared__ float norms[kC];
  const int tid = threadIdx.x, lane = tid & 31, l15 = lane & 15;
  const int hi = (lane >> 4) & 1;
  const int mt = __builtin_amdgcn_readfirstlane(tid >> 5);
  const int arow = c0 + mt * 16 + l15;

#if PERF_HAVE_TDM
  if (mt == 0)  // async DMA of V chunk, overlapped with Phase 1 WMMA work
    tdm_load_2d_bf16(v_bh + (size_t)c0 * kHD, lds_offset_of(vbuf), kHD, kC, kHD);
#else
  for (int i = tid; i < kC * kHD; i += 256)
    vbuf[i] = ((const unsigned short*)(v_bh + (size_t)c0 * kHD))[i];
#endif

  // Phase 1: S = Q' @ K'^T, causal-masked, to LDS (triangular tiles only)
  {
    v8f sacc[8] = {};
    for (int k0 = 0; k0 < kFP; k0 += 32) {
      v16bf a = frag_a(qp_bh, kFP, arow, k0, lane);
#pragma unroll
      for (int g = 0; g < 2; ++g) {
        v16bf bb[4];
#pragma unroll
        for (int q = 0; q < 4; ++q) {
          int nt = g * 4 + q;
          if (nt <= mt)
            bb[q] = frag_b(kp_bh, kFP, c0 + nt * 16 + l15, k0, lane);
        }
#pragma unroll
        for (int q = 0; q < 4; ++q) {
          int nt = g * 4 + q;
          if (nt <= mt) sacc[nt] = wmma_bf(a, bb[q], sacc[nt]);
        }
      }
    }
#pragma unroll
    for (int nt = 0; nt < 8; ++nt)
#pragma unroll
      for (int r = 0; r < 8; ++r) {
        int sq = mt * 16 + r + 8 * hi, sk = nt * 16 + l15;
        float vv = (nt <= mt && sq >= sk) ? sacc[nt][r] : 0.f;
        sS[sq][sk] = (bf16)vv;
      }
  }
#if PERF_HAVE_TDM
  if (mt == 0) tdm_wait0();
#endif
  __syncthreads();
  transpose_v_inplace(vbuf, tid);            // vbuf now V^T [64][128]
  const bf16* vT = (const bf16*)vbuf;

  // Phase 2: acc = Q' @ KVpre  +  S_masked @ V   (16x64 per wave)
  v8f acc[4] = {};
  for (int k0 = 0; k0 < kFP; k0 += 32) {
    v16bf a = frag_a(qp_bh, kFP, arow, k0, lane);
    v16bf bb[4];
#pragma unroll
    for (int nt = 0; nt < 4; ++nt)
      bb[nt] = frag_b(KVt, kFP, nt * 16 + l15, k0, lane);
#pragma unroll
    for (int nt = 0; nt < 4; ++nt) acc[nt] = wmma_bf(a, bb[nt], acc[nt]);
  }
  for (int k0 = 0; k0 < kC; k0 += 32) {
    v16bf a = frag_a(&sS[0][0], kC + 8, mt * 16 + l15, k0, lane);
    v16bf bb[4];
#pragma unroll
    for (int nt = 0; nt < 4; ++nt)
      bb[nt] = frag_b(vT, kC, nt * 16 + l15, k0, lane);
#pragma unroll
    for (int nt = 0; nt < 4; ++nt) acc[nt] = wmma_bf(a, bb[nt], acc[nt]);
  }

  // Phase 3: norms[s] = rowsum(S_masked) + Q'[s] . ksum_prefix
  if (tid < kC) {
    float ns = 0.f;
    for (int k = 0; k < kC; ++k) ns += (float)sS[tid][k];
    const bf16* qrow = qp_bh + (size_t)(c0 + tid) * kFP;
    for (int f = 0; f < kFP; ++f) ns += (float)qrow[f] * ksc[f];
    norms[tid] = ns;
  }
  __syncthreads();

  // Phase 4: normalize and store [B][S][H*HD] bf16 (GEMM-ready layout)
  bf16* attn_b = attn + (size_t)b * kS * kD;
#pragma unroll
  for (int nt = 0; nt < 4; ++nt)
#pragma unroll
    for (int r = 0; r < 8; ++r) {
      int sl = mt * 16 + r + 8 * hi;
      float vv = acc[nt][r] / (norms[sl] + kEps);
      attn_b[(size_t)(c0 + sl) * kD + h * kHD + nt * 16 + l15] = (bf16)vv;
    }
}

// --------------------------- output GEMM (fp32 out) ------------------------
__global__ __launch_bounds__(256) void perf_gemm_f32(
    const bf16* __restrict__ A, const bf16* __restrict__ Bt,
    float* __restrict__ outp, int N, int K) {
  const int tid = threadIdx.x, lane = tid & 31, l15 = lane & 15;
  const int hi = (lane >> 4) & 1;
  const int w = __builtin_amdgcn_readfirstlane(tid >> 5);
  const int m0 = blockIdx.x * 64 + (w & 3) * 16;
  const int nb = blockIdx.y * 128 + (w >> 2) * 64;
  const int arow = m0 + l15;
  v8f acc[4] = {};
  for (int k0 = 0; k0 < K; k0 += 32) {
    __builtin_prefetch(A + (size_t)arow * K + k0 + 128, 0, 1);
    v16bf a = frag_a(A, K, arow, k0, lane);
    v16bf bb[4];
#pragma unroll
    for (int nt = 0; nt < 4; ++nt)
      bb[nt] = frag_b(Bt, K, nb + nt * 16 + l15, k0, lane);
#pragma unroll
    for (int nt = 0; nt < 4; ++nt) acc[nt] = wmma_bf(a, bb[nt], acc[nt]);
  }
#pragma unroll
  for (int nt = 0; nt < 4; ++nt)
#pragma unroll
    for (int r = 0; r < 8; ++r)
      outp[(size_t)(m0 + r + 8 * hi) * N + nb + nt * 16 + l15] = acc[nt][r];
}

// ------------------------------- host --------------------------------------

extern "C" void kernel_launch(void* const* d_in, const int* in_sizes, int n_in,
                              void* d_out, int out_size, void* d_ws,
                              size_t ws_size, hipStream_t stream) {
  const float* x = (const float*)d_in[0];
  const float* Wq = (const float*)d_in[1];
  const float* Wk = (const float*)d_in[2];
  const float* Wv = (const float*)d_in[3];
  const float* Wo = (const float*)d_in[4];
  const float* P = (const float*)d_in[5];
  float* out = (float*)d_out;

  size_t off = 0;
  auto alloc = [&](size_t bytes) -> void* {
    void* p = (char*)d_ws + off;
    off += (bytes + 255) & ~(size_t)255;
    return p;
  };
  bf16* xbf = (bf16*)alloc((size_t)kM * kD * 2);
  bf16* WQPt = (bf16*)alloc((size_t)kH * kFP * kD * 2);
  bf16* WKPt = (bf16*)alloc((size_t)kH * kFP * kD * 2);
  bf16* WvT = (bf16*)alloc((size_t)kD * kD * 2);
  bf16* WoT = (bf16*)alloc((size_t)kD * kD * 2);
  bf16* qp = (bf16*)alloc((size_t)kB * kH * kS * kFP * 2);
  bf16* kpb = (bf16*)alloc((size_t)kB * kH * kS * kFP * 2);
  bf16* vbuf = (bf16*)alloc((size_t)kB * kH * kS * kHD * 2);
  bf16* KVloc = (bf16*)alloc((size_t)kB * kH * kNC * kFP * kHD * 2);
  bf16* KVpreT = (bf16*)alloc((size_t)kB * kH * kNC * kFP * kHD * 2);
  float* ksl = (float*)alloc((size_t)kB * kH * kNC * kFP * 4);
  float* ksp = (float*)alloc((size_t)kB * kH * kNC * kFP * 4);
  bf16* attn = (bf16*)alloc((size_t)kM * kD * 2);

  perf_cvt_bf16<<<(kM * kD + 255) / 256, 256, 0, stream>>>(x, xbf, kM * kD);
  perf_transpose_cvt<<<(kD * kD + 255) / 256, 256, 0, stream>>>(Wv, WvT);
  perf_transpose_cvt<<<(kD * kD + 255) / 256, 256, 0, stream>>>(Wo, WoT);
  perf_fuse_proj<<<(kH * kFP * kD + 255) / 256, 256, 0, stream>>>(Wq, P, WQPt,
                                                                  kScale);
  perf_fuse_proj<<<(kH * kFP * kD + 255) / 256, 256, 0, stream>>>(Wk, P, WKPt,
                                                                  1.0f);
  dim3 gq(kM / 64, (kFP + 127) / 128, kH);
  perf_proj_gemm<<<gq, 256, 0, stream>>>(xbf, WQPt, kD, (long)kFP * kD, 0, qp,
                                         kFP, kFP, kD, 1);
  perf_proj_gemm<<<gq, 256, 0, stream>>>(xbf, WKPt, kD, (long)kFP * kD, 0, kpb,
                                         kFP, kFP, kD, 1);
  dim3 gv(kM / 64, 1, kH);
  perf_proj_gemm<<<gv, 256, 0, stream>>>(xbf, WvT, kD, 0L, kHD, vbuf, kHD, kHD,
                                         kD, 0);
  dim3 gc(kNC, kH, kB);
  perf_chunk_kv<<<gc, 256, 0, stream>>>(kpb, vbuf, KVloc, ksl);
  perf_prefix_kv<<<(kB * kH * kFP * kHD + 255) / 256, 256, 0, stream>>>(KVloc,
                                                                        KVpreT);
  perf_prefix_ksum<<<(kB * kH * kFP + 255) / 256, 256, 0, stream>>>(ksl, ksp);
  perf_intra<<<gc, 256, 0, stream>>>(qp, kpb, vbuf, KVpreT, ksp, attn);
  dim3 go(kM / 64, kD / 128, 1);
  perf_gemm_f32<<<go, 256, 0, stream>>>(attn, WoT, out, kD, kD);
}
